// MultiHeadAttention_37245956391315
// MI455X (gfx1250) — matrix operand order repaired and checked by reading, not compiled, not run on hardware
//
#include <hip/hip_runtime.h>
#include <hip/hip_bf16.h>
#include <cstdint>

typedef __bf16 bf16_t;
typedef __bf16 v16bf __attribute__((ext_vector_type(16)));
typedef __bf16 v8bf  __attribute__((ext_vector_type(8)));
typedef float  v8f   __attribute__((ext_vector_type(8)));
typedef int    v4i   __attribute__((ext_vector_type(4)));

#if defined(__AMDGCN__) && __has_builtin(__builtin_amdgcn_global_load_async_to_lds_b128)
#define USE_ASYNC_LDS 1
#define AS1 __attribute__((address_space(1)))
#define AS3 __attribute__((address_space(3)))
#else
#define USE_ASYNC_LDS 0
#endif

// ---------------------------------------------------------------------------
// WMMA helper: D = A(16x32 bf16) x B(32x16 bf16) + C(16x16 f32)
// ---------------------------------------------------------------------------
__device__ __forceinline__ v8f wmma_bf16(v16bf a, v16bf b, v8f c) {
  return __builtin_amdgcn_wmma_f32_16x16x32_bf16(
      /*neg_a=*/false, a, /*neg_b=*/false, b,
      /*c_mod=*/(short)0, c, /*reuse_a=*/false, /*reuse_b=*/false);
}

// A fragment (16x32, 16-bit): lane l<16 -> M=l, K = k0+{0..7, 16..23}
//                             lane l>=16 -> M=l-16, K = k0+{8..15, 24..31}
__device__ __forceinline__ v16bf load_frag_a(const bf16_t* base, int ld, int m0, int k0) {
  int l   = threadIdx.x & 31;
  int hlf = l >> 4, mr = l & 15;
  const bf16_t* p = base + (size_t)(m0 + mr) * ld + (k0 + 8 * hlf);
  v8bf lo = *(const v8bf*)p;
  v8bf hi = *(const v8bf*)(p + 16);
  return __builtin_shufflevector(lo, hi, 0,1,2,3,4,5,6,7,8,9,10,11,12,13,14,15);
}

// B fragment (32x16, 16-bit): lane holds N = n0 + (l&15), K = k0 + 8*(l>>4) + {0..7, 16..23}
// Source matrix is row-major [N, K] (rows are frag-N, contiguous in frag-K).
__device__ __forceinline__ v16bf load_frag_b(const bf16_t* base, int ld, int n0, int k0) {
  int l  = threadIdx.x & 31;
  int nr = l & 15;
  const bf16_t* p = base + (size_t)(n0 + nr) * ld + (k0 + 8 * (l >> 4));
  v8bf lo = *(const v8bf*)p;
  v8bf hi = *(const v8bf*)(p + 16);
  return __builtin_shufflevector(lo, hi, 0,1,2,3,4,5,6,7,8,9,10,11,12,13,14,15);
}

// ---------------------------------------------------------------------------
// f32 -> bf16 conversion (grid-stride)
// ---------------------------------------------------------------------------
__global__ __launch_bounds__(256) void cvt_f32_bf16(const float* __restrict__ src,
                                                    bf16_t* __restrict__ dst, int n) {
  int i = blockIdx.x * blockDim.x + threadIdx.x;
  int stride = gridDim.x * blockDim.x;
  for (; i < n; i += stride) dst[i] = (bf16_t)src[i];
}

// ---------------------------------------------------------------------------
// Software-pipelined NT WMMA GEMM body (shared by both projection kernels):
//   acc[4][2] += A[m_wave..+64, :] x B[n_wave..+32, :]^T, K-chunks of 32,
//   double-buffered fragments (ping-pong, unrolled x2, no register copies).
// ---------------------------------------------------------------------------
#define GEMM_MAIN_LOOP(Abase, Bbase, K)                                        \
  do {                                                                         \
    v16bf a0[4], b0[2], a1[4], b1[2];                                          \
    _Pragma("unroll") for (int mi = 0; mi < 4; ++mi)                           \
        a0[mi] = load_frag_a(Abase, K, m_wave + 16 * mi, 0);                   \
    _Pragma("unroll") for (int ni = 0; ni < 2; ++ni)                           \
        b0[ni] = load_frag_b(Bbase, K, n_wave + 16 * ni, 0);                   \
    int lr_ = threadIdx.x & 31;                                                \
    for (int k0 = 0; k0 < K; k0 += 64) {                                       \
      __builtin_prefetch(Abase + (size_t)(m_wave + 2 * lr_) * K + k0 + 128, 0, 1); \
      __builtin_prefetch(Bbase + (size_t)(n_wave + lr_) * K + k0 + 128, 0, 1); \
      _Pragma("unroll") for (int mi = 0; mi < 4; ++mi)                         \
          a1[mi] = load_frag_a(Abase, K, m_wave + 16 * mi, k0 + 32);           \
      _Pragma("unroll") for (int ni = 0; ni < 2; ++ni)                         \
          b1[ni] = load_frag_b(Bbase, K, n_wave + 16 * ni, k0 + 32);           \
      _Pragma("unroll") for (int mi = 0; mi < 4; ++mi)                         \
          _Pragma("unroll") for (int ni = 0; ni < 2; ++ni)                     \
              acc[mi][ni] = wmma_bf16(a0[mi], b0[ni], acc[mi][ni]);            \
      int k2 = (k0 + 64 < K) ? k0 + 64 : 0; /* dummy reload on last iter */    \
      _Pragma("unroll") for (int mi = 0; mi < 4; ++mi)                         \
          a0[mi] = load_frag_a(Abase, K, m_wave + 16 * mi, k2);                \
      _Pragma("unroll") for (int ni = 0; ni < 2; ++ni)                         \
          b0[ni] = load_frag_b(Bbase, K, n_wave + 16 * ni, k2);                \
      _Pragma("unroll") for (int mi = 0; mi < 4; ++mi)                         \
          _Pragma("unroll") for (int ni = 0; ni < 2; ++ni)                     \
              acc[mi][ni] = wmma_bf16(a1[mi], b1[ni], acc[mi][ni]);            \
    }                                                                          \
  } while (0)

// ---------------------------------------------------------------------------
// GEMM1: QKV = X[8192,1024] @ Wqkv[3072,1024]^T + b_qkv
// Output scattered to [3][B][H][T][DH] bf16, Q part pre-scaled by 1/sqrt(DH).
// ---------------------------------------------------------------------------
__global__ __launch_bounds__(256) void gemm_qkv_kernel(
    const bf16_t* __restrict__ X, const bf16_t* __restrict__ W,
    const float* __restrict__ bias, bf16_t* __restrict__ QKV) {
  const int K = 1024, T = 2048, H = 16, DH = 64;
  int wid = threadIdx.x >> 5;
  int wm = wid >> 2, wn = wid & 3;
  int m_wave = blockIdx.y * 128 + wm * 64;
  int n_wave = blockIdx.x * 128 + wn * 32;

  v8f acc[4][2];
  v8f z = {0.f, 0.f, 0.f, 0.f, 0.f, 0.f, 0.f, 0.f};
#pragma unroll
  for (int mi = 0; mi < 4; ++mi)
#pragma unroll
    for (int ni = 0; ni < 2; ++ni) acc[mi][ni] = z;

  GEMM_MAIN_LOOP(X, W, K);

  int l = threadIdx.x & 31;
  int hlf = l >> 4, lr = l & 15;
  const size_t HS = (size_t)4 * H * T * DH;  // one of Q/K/V: B*H*T*DH
#pragma unroll
  for (int ni = 0; ni < 2; ++ni) {
    int n = n_wave + ni * 16 + lr;
    float bv = bias[n];
    int which = n >> 10;
    int rem = n & 1023;
    int h = rem >> 6, dh = rem & 63;
    float scl = (which == 0) ? 0.125f : 1.0f;  // fold 1/sqrt(64) into Q
#pragma unroll
    for (int mi = 0; mi < 4; ++mi) {
#pragma unroll
      for (int e = 0; e < 8; ++e) {
        int m = m_wave + mi * 16 + e + 8 * hlf;
        int bb = m >> 11, t = m & 2047;
        size_t idx = (size_t)which * HS +
                     ((((size_t)bb * H + h) * T + t) * DH + dh);
        QKV[idx] = (bf16_t)((acc[mi][ni][e] + bv) * scl);
      }
    }
  }
}

// ---------------------------------------------------------------------------
// Flash attention: 1 workgroup = (b,h) x 64 q-rows; 4 waves x 16 q-rows each.
// kv tiles of 32; K tile staged to LDS (async DMA when available), V tile
// transposed into LDS; S and P*V via bf16 WMMA, online softmax in f32.
// ---------------------------------------------------------------------------
__global__ __launch_bounds__(128) void attn_kernel(const bf16_t* __restrict__ QKV,
                                                   bf16_t* __restrict__ attn16) {
  const int H = 16, T = 2048, DH = 64;
  const size_t QKVsz = (size_t)4 * H * T * DH;

  int bh    = blockIdx.x >> 5;   // 0..63
  int qtile = blockIdx.x & 31;   // 0..31
  int tid = threadIdx.x;
  int w = tid >> 5;
  int l = tid & 31;
  int hlf = l >> 4, lr = l & 15;

  const bf16_t* Qp = QKV;
  const bf16_t* Kp = QKV + QKVsz;
  const bf16_t* Vp = QKV + 2 * QKVsz;

  int q0 = qtile * 64 + w * 16;
  int rowbase = bh * T;

  __shared__ bf16_t Kl[32][72];      // K tile: [n][d], 144B row stride (16B-aligned)
  __shared__ bf16_t Vt[64][40];      // V^T tile: [d][n], 80B row stride
  __shared__ bf16_t Pt[4][16][40];   // per-wave P tile: [qrow][n]

  // Q fragments for this wave's 16 rows (K-dim 64 = two chunks of 32)
  v16bf aq[2];
#pragma unroll
  for (int c = 0; c < 2; ++c) aq[c] = load_frag_a(Qp, DH, rowbase + q0, 32 * c);

  v8f z = {0.f, 0.f, 0.f, 0.f, 0.f, 0.f, 0.f, 0.f};
  v8f o[4];
#pragma unroll
  for (int j = 0; j < 4; ++j) o[j] = z;
  float mrun[8], lrun[8];
#pragma unroll
  for (int i = 0; i < 8; ++i) { mrun[i] = -3.0e38f; lrun[i] = 0.f; }

  // per-thread staging coordinates: each of 128 threads moves 16 bf16 (32B)
  int sn = tid >> 2;               // 0..31 (kv row)
  int sd = (tid & 3) * 16;         // 0,16,32,48 (d column base)

  for (int n0 = 0; n0 < T; n0 += 32) {
    __syncthreads();  // previous tiles fully consumed

    // ---- stage K tile (row-major) into LDS ----
    const bf16_t* kp = Kp + ((size_t)(rowbase + n0 + sn)) * DH + sd;
#if USE_ASYNC_LDS
    __builtin_amdgcn_global_load_async_to_lds_b128(
        (AS1 v4i*)(uintptr_t)kp, (AS3 v4i*)(uintptr_t)&Kl[sn][sd], 0, 0);
    __builtin_amdgcn_global_load_async_to_lds_b128(
        (AS1 v4i*)(uintptr_t)(kp + 8), (AS3 v4i*)(uintptr_t)&Kl[sn][sd + 8], 0, 0);
#else
    {
      v8bf k0v = *(const v8bf*)kp;
      v8bf k1v = *(const v8bf*)(kp + 8);
      *(v8bf*)&Kl[sn][sd] = k0v;
      *(v8bf*)&Kl[sn][sd + 8] = k1v;
    }
#endif

    // ---- stage V^T tile into LDS (manual transpose) ----
    {
      const bf16_t* vp = Vp + ((size_t)(rowbase + n0 + sn)) * DH + sd;
      v16bf vv = *(const v16bf*)vp;
#pragma unroll
      for (int j = 0; j < 16; ++j) Vt[sd + j][sn] = vv[j];
    }
#if USE_ASYNC_LDS
    asm volatile("s_wait_asynccnt 0x0" ::: "memory");
#endif
    __syncthreads();

    // S = Q @ K^T  (16 x 32 tile, N split into two 16-col accumulators)
    v8f s[2];
    s[0] = z; s[1] = z;
#pragma unroll
    for (int c = 0; c < 2; ++c) {
#pragma unroll
      for (int nc = 0; nc < 2; ++nc) {
        v16bf bk = load_frag_b(&Kl[0][0], 72, nc * 16, 32 * c);
        s[nc] = wmma_bf16(aq[c], bk, s[nc]);
      }
    }

    // online softmax update (row = e + 8*hlf; row's 16 N-values across lanes)
    float scale[8];
#pragma unroll
    for (int i = 0; i < 8; ++i) {
      float mx = fmaxf(s[0][i], s[1][i]);
#pragma unroll
      for (int off = 1; off < 16; off <<= 1) mx = fmaxf(mx, __shfl_xor(mx, off, 32));
      float mn = fmaxf(mrun[i], mx);
      float sc = __expf(mrun[i] - mn);
      float p0 = __expf(s[0][i] - mn);
      float p1 = __expf(s[1][i] - mn);
      s[0][i] = p0; s[1][i] = p1;
      float rs = p0 + p1;
#pragma unroll
      for (int off = 1; off < 16; off <<= 1) rs += __shfl_xor(rs, off, 32);
      lrun[i] = lrun[i] * sc + rs;
      mrun[i] = mn;
      scale[i] = sc;
    }
#pragma unroll
    for (int j = 0; j < 4; ++j)
#pragma unroll
      for (int i = 0; i < 8; ++i) o[j][i] *= scale[i];

    // P (C-layout f32) -> LDS -> A-layout bf16 fragment
#pragma unroll
    for (int nc = 0; nc < 2; ++nc)
#pragma unroll
      for (int i = 0; i < 8; ++i)
        Pt[w][i + 8 * hlf][nc * 16 + lr] = (bf16_t)s[nc][i];
    asm volatile("s_wait_dscnt 0" ::: "memory");
    v16bf ap = load_frag_a(&Pt[w][0][0], 40, 0, 0);

    // O += P(16x32) @ V(32x64): 4 B-fragments from Vt (rows = d, contiguous n)
#pragma unroll
    for (int j = 0; j < 4; ++j) {
      v16bf bv = load_frag_b(&Vt[0][0], 40, j * 16, 0);
      o[j] = wmma_bf16(ap, bv, o[j]);
    }
  }

  // normalize and store to [B,T,H*DH] bf16
  float inv[8];
#pragma unroll
  for (int i = 0; i < 8; ++i) inv[i] = 1.0f / lrun[i];
  int b_ = bh >> 4, h = bh & 15;
#pragma unroll
  for (int j = 0; j < 4; ++j) {
    int d = j * 16 + lr;
#pragma unroll
    for (int i = 0; i < 8; ++i) {
      int q = q0 + i + 8 * hlf;
      attn16[((size_t)(b_ * T + q)) * 1024 + h * DH + d] = (bf16_t)(o[j][i] * inv[i]);
    }
  }
}

// ---------------------------------------------------------------------------
// GEMM2: out[8192,1024] = ATT[8192,1024] @ Wout[1024,1024]^T + b_out  (f32 out)
// ---------------------------------------------------------------------------
__global__ __launch_bounds__(256) void gemm_out_kernel(
    const bf16_t* __restrict__ A, const bf16_t* __restrict__ W,
    const float* __restrict__ bias, float* __restrict__ out) {
  const int K = 1024, N = 1024;
  int wid = threadIdx.x >> 5;
  int wm = wid >> 2, wn = wid & 3;
  int m_wave = blockIdx.y * 128 + wm * 64;
  int n_wave = blockIdx.x * 128 + wn * 32;

  v8f acc[4][2];
  v8f z = {0.f, 0.f, 0.f, 0.f, 0.f, 0.f, 0.f, 0.f};
#pragma unroll
  for (int mi = 0; mi < 4; ++mi)
#pragma unroll
    for (int ni = 0; ni < 2; ++ni) acc[mi][ni] = z;

  GEMM_MAIN_LOOP(A, W, K);

  int l = threadIdx.x & 31;
  int hlf = l >> 4, lr = l & 15;
#pragma unroll
  for (int ni = 0; ni < 2; ++ni) {
    int n = n_wave + ni * 16 + lr;
    float bv = bias[n];
#pragma unroll
    for (int mi = 0; mi < 4; ++mi) {
#pragma unroll
      for (int e = 0; e < 8; ++e) {
        int m = m_wave + mi * 16 + e + 8 * hlf;
        out[(size_t)m * N + n] = acc[mi][ni][e] + bv;
      }
    }
  }
}

// ---------------------------------------------------------------------------
// Launch
// ---------------------------------------------------------------------------
extern "C" void kernel_launch(void* const* d_in, const int* in_sizes, int n_in,
                              void* d_out, int out_size, void* d_ws, size_t ws_size,
                              hipStream_t stream) {
  const float* x     = (const float*)d_in[0];  // [4,2048,1024]
  const float* w_qkv = (const float*)d_in[1];  // [3072,1024]
  const float* b_qkv = (const float*)d_in[2];  // [3072]
  const float* w_out = (const float*)d_in[3];  // [1024,1024]
  const float* b_out = (const float*)d_in[4];  // [1024]
  float* out = (float*)d_out;

  const size_t nX  = (size_t)8192 * 1024;
  const size_t nW1 = (size_t)3072 * 1024;
  const size_t nW2 = (size_t)1024 * 1024;
  const size_t nHS = (size_t)4 * 16 * 2048 * 64;  // B*H*T*DH

  bf16_t* X16  = (bf16_t*)d_ws;
  bf16_t* W1   = X16 + nX;
  bf16_t* W2   = W1 + nW1;
  bf16_t* QKVb = W2 + nW2;
  bf16_t* ATT  = QKVb + 3 * nHS;

  cvt_f32_bf16<<<2048, 256, 0, stream>>>(x, X16, (int)nX);
  cvt_f32_bf16<<<2048, 256, 0, stream>>>(w_qkv, W1, (int)nW1);
  cvt_f32_bf16<<<2048, 256, 0, stream>>>(w_out, W2, (int)nW2);

  gemm_qkv_kernel<<<dim3(24, 64), 256, 0, stream>>>(X16, W1, b_qkv, QKVb);
  attn_kernel<<<2048, 128, 0, stream>>>(QKVb, ATT);
  gemm_out_kernel<<<dim3(8, 64), 256, 0, stream>>>(ATT, W2, b_out, out);
}
